// WRPBlockLinear_40991167873569
// MI455X (gfx1250) — matrix operand order repaired and checked by reading, not compiled, not run on hardware
//
// WRPBlockLinear for MI455X (gfx1250): 2:4 sparse SWMMAC + split-precision bf16.
//
// y[s,o] = x @ W_semi.T + einsum(gathered x blocks, recover_values) + bias
// S = O = K = 4096, BS = 32, TOPN = 8, fp32 in/out.
//
// W_semi is exactly 2-of-4 sparse along K (reference keeps top-2 |w| per group
// of 4), i.e. the native input format of V_SWMMAC_F32_16X16X64_BF16. We swap
// GEMM roles so W is the sparse A operand: D[m=o][n=s] = W[o,:] . x[s,:].
//   * Pack W into half-size bf16 hi/lo values + 2-bit index words (ISA 7.12.4).
//     Packed A uses the standard dense 16-bit 16x32 A register layout (7.12.2).
//   * Split x and RV into bf16 hi + residual-lo; keep hi*hi + hi*lo + lo*hi
//     (~1e-5 rel err) -> 3 SWMMACs per 16x16x64 tile: half the matrix ops of
//     the dense 16x16x32 path for the dominant term.
//   * Recovery = 8 gathered K=32 steps per 32-row output block with dense
//     V_WMMA_F32_16X16X32_BF16 into the same accumulators; bias in epilogue.
//   * Role swap makes the epilogue o-contiguous per lane: b128 stores.
//
// Workspace (102 MB): xh/xl 64MB, packed W hi/lo 32MB, sparse idx 2MB, RV hi/lo 4MB.

#include <hip/hip_runtime.h>

typedef __attribute__((ext_vector_type(32))) __bf16 v32bf;
typedef __attribute__((ext_vector_type(16))) __bf16 v16bf;
typedef __attribute__((ext_vector_type(8)))  __bf16 v8bf;
typedef __attribute__((ext_vector_type(8)))  float  v8f;

constexpr int S_DIM  = 4096;
constexpr int K_DIM  = 4096;
constexpr int O_DIM  = 4096;
constexpr int BSZ    = 32;
constexpr int TOPN_N = 8;
constexpr int KB64   = K_DIM / 64;   // 64 K-blocks of 64
constexpr int PK     = K_DIM / 2;    // packed row length (2048)

// ---------------------------------------------------------------------------
// f32 -> (bf16 hi, bf16 lo) split.
// ---------------------------------------------------------------------------
__global__ void split_bf16_kernel(const float* __restrict__ src,
                                  __bf16* __restrict__ hi,
                                  __bf16* __restrict__ lo,
                                  int n) {
  int i = blockIdx.x * blockDim.x + threadIdx.x;
  if (i < n) {
    float v  = src[i];
    __bf16 h = (__bf16)v;
    hi[i] = h;
    lo[i] = (__bf16)(v - (float)h);
  }
}

// ---------------------------------------------------------------------------
// 2:4 pack of W: one thread per (o, K64-block). Emits 32 packed bf16 hi/lo
// values (packed col 2g+e for group g, kept element e) and two 32-bit index
// words (lane-half 0: groups 0-7 / cols 0-31, lane-half 1: groups 8-15 /
// cols 32-63; 4 bits per group, idx0 < idx1).  (ISA 7.12.4)
// ---------------------------------------------------------------------------
__global__ void pack_sparse_kernel(const float* __restrict__ W,
                                   __bf16* __restrict__ wph,
                                   __bf16* __restrict__ wpl,
                                   unsigned int* __restrict__ sidx) {
  int tid = blockIdx.x * blockDim.x + threadIdx.x;   // o * KB64 + kb
  if (tid >= O_DIM * KB64) return;
  const int o  = tid >> 6;
  const int kb = tid & (KB64 - 1);

  const float* wr = W + (size_t)o * K_DIM + kb * 64;
  __bf16* ph = wph + (size_t)o * PK + kb * 32;
  __bf16* pl = wpl + (size_t)o * PK + kb * 32;

  unsigned int code0 = 0, code1 = 0;
#pragma unroll
  for (int g = 0; g < 16; ++g) {
    float a[4], ab[4];
#pragma unroll
    for (int i = 0; i < 4; ++i) { a[i] = wr[4 * g + i]; ab[i] = fabsf(a[i]); }
    // top-2 by |w| (reference: |w| >= 2nd-largest |w|), indices ascending
    int i0 = -1, i1 = -1;
#pragma unroll
    for (int i = 0; i < 4; ++i) {
      int cnt = 0;
#pragma unroll
      for (int j = 0; j < 4; ++j)
        cnt += ((ab[j] > ab[i]) || (ab[j] == ab[i] && j < i)) ? 1 : 0;
      if (cnt < 2) { if (i0 < 0) i0 = i; else i1 = i; }
    }
    const float v0 = a[i0], v1 = a[i1];
    const __bf16 h0 = (__bf16)v0, h1 = (__bf16)v1;
    ph[2 * g]     = h0;  pl[2 * g]     = (__bf16)(v0 - (float)h0);
    ph[2 * g + 1] = h1;  pl[2 * g + 1] = (__bf16)(v1 - (float)h1);
    const unsigned int c = (unsigned int)(i0 | (i1 << 2));
    if (g < 8) code0 |= c << (4 * g);
    else       code1 |= c << (4 * (g - 8));
  }
  sidx[(size_t)tid * 2]     = code0;   // lane half 0 (orig cols 0-31 of block)
  sidx[(size_t)tid * 2 + 1] = code1;   // lane half 1 (orig cols 32-63 of block)
}

// ---------------------------------------------------------------------------
// Fragment loaders (wave32).
// ---------------------------------------------------------------------------
// Dense 16-bit A 16x32 (interleaved-by-8, ISA 7.12.2): lane half h holds K
// {8h..8h+7, 16+8h..16+8h+7} of row `row`. Also used for the *packed* sparse
// A (stored exactly as a dense half-size 16x32 A matrix).
__device__ __forceinline__ v16bf load_frag_a32(const __bf16* __restrict__ base,
                                               int row, int ld, int k0, int hsel) {
  const __bf16* p = base + (size_t)row * ld + k0 + hsel * 8;
  union { v16bf v; v8bf h[2]; } u;
  u.h[0] = *(const v8bf*)(p);
  u.h[1] = *(const v8bf*)(p + 16);
  return u.v;
}

// Dense B 32x16: lane half h holds K {8h..8h+7, 16+8h..16+8h+7} of column `col`.
__device__ __forceinline__ v16bf load_frag_b32(const __bf16* __restrict__ base,
                                               int col, int ld, int k0, int hsel) {
  const __bf16* p = base + (size_t)col * ld + k0 + hsel * 8;
  union { v16bf v; v8bf h[2]; } u;
  u.h[0] = *(const v8bf*)(p);
  u.h[1] = *(const v8bf*)(p + 16);
  return u.v;
}

// Dense B 64x16 (ISA 7.12.4): VGPRs 0-7 = K{0-15|16-31 by lane half},
// VGPRs 8-15 = K{32-47|48-63 by lane half}, of column `col`.
__device__ __forceinline__ v32bf load_frag_b64(const __bf16* __restrict__ base,
                                               int col, int ld, int k0, int hsel) {
  const __bf16* p = base + (size_t)col * ld + k0 + hsel * 16;
  union { v32bf v; v8bf h[4]; } u;
  u.h[0] = *(const v8bf*)(p);
  u.h[1] = *(const v8bf*)(p + 8);
  u.h[2] = *(const v8bf*)(p + 32);
  u.h[3] = *(const v8bf*)(p + 40);
  return u.v;
}

__device__ __forceinline__ v8f wmma_bf16(v16bf a, v16bf b, v8f c) {
  return __builtin_amdgcn_wmma_f32_16x16x32_bf16(false, a, false, b,
                                                 (short)0, c, false, false);
}

__device__ __forceinline__ v8f swmmac_bf16(v16bf a, v32bf b, v8f c, int idx) {
  // (neg_a, A, neg_b, B, C, sparse_index, reuse_a, reuse_b)
  return __builtin_amdgcn_swmmac_f32_16x16x64_bf16(false, a, false, b,
                                                   c, idx, false, false);
}

// ---------------------------------------------------------------------------
// Fused kernel: sparse semi GEMM (SWMMAC) + recovery (WMMA) + bias.
// Roles: M = output dim o (W rows), N = sequence dim s (x rows).
// grid = (S_DIM/128, O_DIM/128), block = 256 (8 waves, wave tile 32M x 64N).
// ---------------------------------------------------------------------------
__global__ __launch_bounds__(256) void wrp_gemm_kernel(
    const __bf16* __restrict__ xh,  const __bf16* __restrict__ xl,
    const __bf16* __restrict__ wph, const __bf16* __restrict__ wpl,
    const unsigned int* __restrict__ sidx,
    const __bf16* __restrict__ rvh, const __bf16* __restrict__ rvl,
    const int*    __restrict__ ridx,
    const float*  __restrict__ bias,
    float*        __restrict__ out) {
  const int lane = threadIdx.x & 31;
  const int wave = threadIdx.x >> 5;
  const int wm   = wave & 3;         // 4 wave rows (M/o)
  const int wn   = wave >> 2;        // 2 wave cols (N/s)
  const int lq   = lane & 15;
  const int hsel = lane >> 4;

  const int m_base = blockIdx.y * 128 + wm * 32;  // o: 2 M tiles of 16
  const int n_base = blockIdx.x * 128 + wn * 64;  // s: 4 N tiles of 16

  v8f acc[2][4];
#pragma unroll
  for (int mt = 0; mt < 2; ++mt)
#pragma unroll
    for (int nt = 0; nt < 4; ++nt)
      acc[mt][nt] = (v8f)(0.0f);

  // ---------------- main loop: 2:4 sparse W  x  dense x, K step 64 --------
  for (int kb = 0; kb < KB64; ++kb) {
    v16bf ah[2], al[2];
    int   sid[2];
#pragma unroll
    for (int mt = 0; mt < 2; ++mt) {
      const int row = m_base + mt * 16 + lq;
      ah[mt]  = load_frag_a32(wph, row, PK, kb * 32, hsel);  // packed cols
      al[mt]  = load_frag_a32(wpl, row, PK, kb * 32, hsel);
      sid[mt] = (int)sidx[((size_t)row * KB64 + kb) * 2 + hsel];
    }
#pragma unroll
    for (int nt = 0; nt < 4; ++nt) {
      const int col = n_base + nt * 16 + lq;      // s row of x
      const v32bf bh = load_frag_b64(xh, col, K_DIM, kb * 64, hsel);
      const v32bf bl = load_frag_b64(xl, col, K_DIM, kb * 64, hsel);
#pragma unroll
      for (int mt = 0; mt < 2; ++mt)
        acc[mt][nt] = swmmac_bf16(ah[mt], bh, acc[mt][nt], sid[mt]);
#pragma unroll
      for (int mt = 0; mt < 2; ++mt)
        acc[mt][nt] = swmmac_bf16(ah[mt], bl, acc[mt][nt], sid[mt]);
#pragma unroll
      for (int mt = 0; mt < 2; ++mt)
        acc[mt][nt] = swmmac_bf16(al[mt], bh, acc[mt][nt], sid[mt]);
    }
  }

  // ---------------- recovery: dense WMMA over gathered blocks -------------
  // Wave's 32 M rows are exactly one 32-wide output block r.
  const int r = m_base >> 5;
  for (int t = 0; t < TOPN_N; ++t) {
    const int kblk = ridx[r * TOPN_N + t] * BSZ;                // gathered K
    const size_t rvoff = (size_t)(r * TOPN_N + t) * BSZ * BSZ;  // RV[r,t] block
    v16bf ah2[2], al2[2], bh2[4], bl2[4];
#pragma unroll
    for (int mt = 0; mt < 2; ++mt) {   // A[m=o_local][k=i] = RV[r,t,o,i]
      ah2[mt] = load_frag_a32(rvh + rvoff, mt * 16 + lq, BSZ, 0, hsel);
      al2[mt] = load_frag_a32(rvl + rvoff, mt * 16 + lq, BSZ, 0, hsel);
    }
#pragma unroll
    for (int nt = 0; nt < 4; ++nt) {   // B[k=i][n=s] = x[s, kblk+i]
      bh2[nt] = load_frag_b32(xh, n_base + nt * 16 + lq, K_DIM, kblk, hsel);
      bl2[nt] = load_frag_b32(xl, n_base + nt * 16 + lq, K_DIM, kblk, hsel);
    }
#pragma unroll
    for (int mt = 0; mt < 2; ++mt)
#pragma unroll
      for (int nt = 0; nt < 4; ++nt)
        acc[mt][nt] = wmma_bf16(ah2[mt], bh2[nt], acc[mt][nt]);
#pragma unroll
    for (int mt = 0; mt < 2; ++mt)
#pragma unroll
      for (int nt = 0; nt < 4; ++nt)
        acc[mt][nt] = wmma_bf16(ah2[mt], bl2[nt], acc[mt][nt]);
#pragma unroll
    for (int mt = 0; mt < 2; ++mt)
#pragma unroll
      for (int nt = 0; nt < 4; ++nt)
        acc[mt][nt] = wmma_bf16(al2[mt], bh2[nt], acc[mt][nt]);
  }

  // ---------------- epilogue: + bias[o], o-contiguous b128 stores ---------
  // D layout: lane (hsel,lq), VGPR j -> M(o) = 8*hsel + j, N(s) = lq.
#pragma unroll
  for (int mt = 0; mt < 2; ++mt) {
    const int o0 = m_base + mt * 16 + hsel * 8;     // 8 consecutive o per lane
    const v8f bv = *(const v8f*)(bias + o0);        // 32B aligned
#pragma unroll
    for (int nt = 0; nt < 4; ++nt) {
      const int s = n_base + nt * 16 + lq;
      const v8f rvec = acc[mt][nt] + bv;
      *(v8f*)(out + (size_t)s * O_DIM + o0) = rvec; // 2x global_store_b128
    }
  }
}

// ---------------------------------------------------------------------------
// Launch. Inputs: x, W_semi, recover_values, recover_indices, bias.
// Workspace layout (needs 102 MB):
//   [0,32M) xh   [32M,64M) xl   [64M,80M) wph   [80M,96M) wpl
//   [96M,98M) sidx   [98M,100M) rvh   [100M,102M) rvl
// ---------------------------------------------------------------------------
extern "C" void kernel_launch(void* const* d_in, const int* in_sizes, int n_in,
                              void* d_out, int out_size, void* d_ws, size_t ws_size,
                              hipStream_t stream) {
  const float* x    = (const float*)d_in[0];
  const float* W    = (const float*)d_in[1];
  const float* RV   = (const float*)d_in[2];
  const int*   ridx = (const int*)d_in[3];
  const float* bias = (const float*)d_in[4];
  float*       out  = (float*)d_out;

  char* ws = (char*)d_ws;
  const size_t XN  = (size_t)S_DIM * K_DIM;                      // 16M elems
  const size_t XB  = XN * sizeof(__bf16);                        // 32 MB
  const size_t PN  = (size_t)O_DIM * PK;                         // 8M elems
  const size_t PB  = PN * sizeof(__bf16);                        // 16 MB
  const size_t SB  = (size_t)O_DIM * KB64 * 2 * sizeof(unsigned);// 2 MB
  const size_t RVN = (size_t)(O_DIM / BSZ) * TOPN_N * BSZ * BSZ; // 1M elems
  const size_t RVB = RVN * sizeof(__bf16);                       // 2 MB

  __bf16*       xh   = (__bf16*)(ws);
  __bf16*       xl   = (__bf16*)(ws + XB);
  __bf16*       wph  = (__bf16*)(ws + 2 * XB);
  __bf16*       wpl  = (__bf16*)(ws + 2 * XB + PB);
  unsigned int* sidx = (unsigned int*)(ws + 2 * XB + 2 * PB);
  __bf16*       rvh  = (__bf16*)(ws + 2 * XB + 2 * PB + SB);
  __bf16*       rvl  = (__bf16*)(ws + 2 * XB + 2 * PB + SB + RVB);

  const int T = 256;
  split_bf16_kernel<<<(int)((XN  + T - 1) / T), T, 0, stream>>>(x,  xh,  xl,  (int)XN);
  split_bf16_kernel<<<(int)((RVN + T - 1) / T), T, 0, stream>>>(RV, rvh, rvl, (int)RVN);
  const int PT = O_DIM * KB64;   // 262144 pack threads
  pack_sparse_kernel<<<(PT + T - 1) / T, T, 0, stream>>>(W, wph, wpl, sidx);

  dim3 grid(S_DIM / 128, O_DIM / 128);
  wrp_gemm_kernel<<<grid, 256, 0, stream>>>(xh, xl, wph, wpl, sidx,
                                            rvh, rvl, ridx, bias, out);
}
